// RouteNetModel_64785286693615
// MI455X (gfx1250) — compile-verified, matrix operand order checked by reading, not run on hardware
//
#include <hip/hip_runtime.h>
#include <hip/hip_bf16.h>

// ---------------------------------------------------------------------------
// Types for CDNA5 WMMA (gfx1250, wave32)
// ---------------------------------------------------------------------------
typedef __bf16 bf16_t;
typedef __bf16 bf16x8  __attribute__((ext_vector_type(8)));
typedef __bf16 bf16x16 __attribute__((ext_vector_type(16)));
typedef float  f32x8   __attribute__((ext_vector_type(8)));

#define ACT_NONE 0
#define ACT_SELU 1
#define ACT_RELU 2

enum {
    N_NODES = 512,
    N_LINKS = 4096,
    N_PATHS = 20000,
    D       = 128,
    HEADS   = 24,
    R       = 256,
    T_STEPS = 3,
    MAX_LEN = 8,
    G3      = 3 * D   // 384, GRU gate width
};

__device__ __forceinline__ bf16_t to_bf16(float x) {
    // round-to-nearest-even f32 -> bf16 via bit math (robust across toolchains)
    unsigned u = __builtin_bit_cast(unsigned, x);
    unsigned r = (u + 0x7FFFu + ((u >> 16) & 1u)) >> 16;
    unsigned short s = (unsigned short)r;
    return __builtin_bit_cast(bf16_t, s);
}

__device__ __forceinline__ float sigmoidf(float x) {
    return 1.0f / (1.0f + __expf(-x));
}

__device__ __forceinline__ bf16x16 frag16(const bf16_t* p) {
    // 16-bit A/B layout: per lane, 8 contiguous elems at K+0, 8 at K+16
    bf16x8 lo = *(const bf16x8*)(p);
    bf16x8 hi = *(const bf16x8*)(p + 16);
    bf16x16 v;
#pragma unroll
    for (int i = 0; i < 8; ++i) { v[i] = lo[i]; v[8 + i] = hi[i]; }
    return v;
}

__device__ __forceinline__ float apply_act(float v, int act) {
    if (act == ACT_SELU) {
        const float sc = 1.0507009873554805f, al = 1.6732632423543772f;
        v = sc * (v > 0.0f ? v : al * (__expf(v) - 1.0f));
    } else if (act == ACT_RELU) {
        v = fmaxf(v, 0.0f);
    }
    return v;
}

// ---------------------------------------------------------------------------
// Generic batched WMMA GEMM, 2x2 register-blocked (32x32 output per wave):
//   C[b] = act(A[b] (MxK bf16 row-major, lda) x B[b]^T stored [N][K] (ldb) + bias)
// store or scaled atomic-add. M, N multiples of 32; K multiple of 32.
// Each k-chunk: 4 fragment loads feed 4 v_wmma_f32_16x16x32_bf16 (2x reuse of
// A and B fragments vs the naive 1-tile/wave form).
// ---------------------------------------------------------------------------
__global__ void wmma_gemm_bf16(const bf16_t* __restrict__ A, long lda, long strideA,
                               const bf16_t* __restrict__ B, long ldb, long strideB,
                               const float* __restrict__ bias,
                               float* __restrict__ C, long ldc, long strideC,
                               int M, int N, int K,
                               int act, int atomicMode, float alpha)
{
    const int wave = threadIdx.x >> 5;
    const int lane = threadIdx.x & 31;
    const int tilesN = N >> 5;             // 32-wide super-tiles
    const int tilesM = M >> 5;
    const int tile = blockIdx.x * (blockDim.x >> 5) + wave;
    if (tile >= tilesM * tilesN) return;   // whole-wave uniform exit (EXEC stays all-1)
    const int tm = tile / tilesN;
    const int tn = tile % tilesN;

    const long batch = blockIdx.z;
    A += batch * strideA;
    B += batch * strideB;
    C += batch * strideC;

    const int hb  = lane >> 4;     // half of wave selects K sub-block
    const int l16 = lane & 15;
    const long rowA0 = (long)(tm * 32 +      l16) * lda + hb * 8;
    const long rowA1 = (long)(tm * 32 + 16 + l16) * lda + hb * 8;
    const long rowB0 = (long)(tn * 32 +      l16) * ldb + hb * 8;
    const long rowB1 = (long)(tn * 32 + 16 + l16) * ldb + hb * 8;

    f32x8 acc00 = {0.f,0.f,0.f,0.f,0.f,0.f,0.f,0.f};
    f32x8 acc01 = acc00, acc10 = acc00, acc11 = acc00;

    for (int kk = 0; kk < K; kk += 32) {
        bf16x16 a0 = frag16(A + rowA0 + kk);
        bf16x16 a1 = frag16(A + rowA1 + kk);
        bf16x16 b0 = frag16(B + rowB0 + kk);
        bf16x16 b1 = frag16(B + rowB1 + kk);
        acc00 = __builtin_amdgcn_wmma_f32_16x16x32_bf16(false, a0, false, b0, (short)0, acc00, false, false);
        acc01 = __builtin_amdgcn_wmma_f32_16x16x32_bf16(false, a0, false, b1, (short)0, acc01, false, false);
        acc10 = __builtin_amdgcn_wmma_f32_16x16x32_bf16(false, a1, false, b0, (short)0, acc10, false, false);
        acc11 = __builtin_amdgcn_wmma_f32_16x16x32_bf16(false, a1, false, b1, (short)0, acc11, false, false);
    }

    const int n0 = tn * 32 + l16;
    const int n1 = n0 + 16;
    const float bs0 = bias ? bias[n0] : 0.0f;
    const float bs1 = bias ? bias[n1] : 0.0f;
    const int m0 = tm * 32 + hb * 8;       // C/D layout: M = r + 8*half (+ sub-tile*16)

#pragma unroll
    for (int r = 0; r < 8; ++r) {
        const long ma = (long)(m0 + r) * ldc;
        const long mb = (long)(m0 + 16 + r) * ldc;
        float v00 = apply_act(acc00[r] + bs0, act);
        float v01 = apply_act(acc01[r] + bs1, act);
        float v10 = apply_act(acc10[r] + bs0, act);
        float v11 = apply_act(acc11[r] + bs1, act);
        if (atomicMode) {
            atomicAdd(&C[ma + n0], v00 * alpha);
            atomicAdd(&C[ma + n1], v01 * alpha);
            atomicAdd(&C[mb + n0], v10 * alpha);
            atomicAdd(&C[mb + n1], v11 * alpha);
        } else {
            C[ma + n0] = v00;
            C[ma + n1] = v01;
            C[mb + n0] = v10;
            C[mb + n1] = v11;
        }
    }
}

// ---------------------------------------------------------------------------
// Elementwise / setup kernels
// ---------------------------------------------------------------------------
__global__ void k_zero_f32(float* p, long n) {
    long i = (long)blockIdx.x * blockDim.x + threadIdx.x;
    if (i < n) p[i] = 0.0f;
}

__global__ void k_f32_to_bf16(const float* __restrict__ s, bf16_t* __restrict__ d, long n) {
    long i = (long)blockIdx.x * blockDim.x + threadIdx.x;
    if (i < n) d[i] = to_bf16(s[i]);
}

// src [K][N] f32 -> dst [N][K] bf16
__global__ void k_transpose_bf16(const float* __restrict__ s, bf16_t* __restrict__ d,
                                 int K, int N) {
    long i = (long)blockIdx.x * blockDim.x + threadIdx.x;
    if (i >= (long)K * N) return;
    int n = (int)(i / K), k = (int)(i % K);
    d[i] = to_bf16(s[(long)k * N + n]);
}

__global__ void k_build_node_state(const float* qp, const float* w1, const float* w2,
                                   const float* w3, const float* qs, float* ns) {
    int i = blockIdx.x * blockDim.x + threadIdx.x;
    if (i >= N_NODES * D) return;
    int n = i >> 7, c = i & 127;
    float v = 0.0f;
    if      (c == 0) v = qp[n];
    else if (c == 1) v = w1[n];
    else if (c == 2) v = w2[n];
    else if (c == 3) v = w3[n];
    else if (c <= 6) v = qs[n * 3 + (c - 4)];
    ns[i] = v;
}

__global__ void k_build_path_state(const float* bw, const float* tos, float* ps) {
    long i = (long)blockIdx.x * blockDim.x + threadIdx.x;
    if (i >= (long)N_PATHS * D) return;
    int p = (int)(i >> 7), c = (int)(i & 127);
    ps[i] = (c == 0) ? bw[p] : (c == 1 ? tos[p] : 0.0f);
}

__global__ void k_reduce_ssq(const float* cap, float* out, int n) {
    __shared__ float red[256];
    float s = 0.0f;
    for (int i = threadIdx.x; i < n; i += 256) s += cap[i] * cap[i];
    red[threadIdx.x] = s; __syncthreads();
    for (int st = 128; st > 0; st >>= 1) {
        if (threadIdx.x < st) red[threadIdx.x] += red[threadIdx.x + st];
        __syncthreads();
    }
    if (threadIdx.x == 0) out[0] = red[0];
}

__global__ void k_scatter_adj(const int* idx, const float* cap, const float* ssq,
                              float* adj, int n) {
    int i = blockIdx.x * blockDim.x + threadIdx.x;
    if (i < n) adj[idx[i]] = cap[i] * rsqrtf(ssq[0]);
}

__global__ void k_count_paths(const int* paths, const int* seqs, const int* nodes,
                              int* lens, int* node_at, int nsteps) {
    int i = blockIdx.x * blockDim.x + threadIdx.x;
    if (i >= nsteps) return;
    int p = paths[i];
    atomicAdd(&lens[p], 1);
    node_at[p * MAX_LEN + seqs[i]] = nodes[i];
}

__global__ void k_zero_i32(int* p, long n) {
    long i = (long)blockIdx.x * blockDim.x + threadIdx.x;
    if (i < n) p[i] = 0;
}

// e_s[n][h], e_n[n][h] from xp (N_NODES x HEADS*D)
__global__ void k_es_en(const float* __restrict__ xp, const float* __restrict__ a_s,
                        const float* __restrict__ a_n, float* e_s, float* e_n) {
    int i = blockIdx.x * blockDim.x + threadIdx.x;
    if (i >= N_NODES * HEADS) return;
    int n = i / HEADS, h = i % HEADS;
    const float* row = xp + (long)n * (HEADS * D) + h * D;
    const float* as = a_s + h * D;
    const float* an = a_n + h * D;
    float ss = 0.0f, sn = 0.0f;
    for (int c = 0; c < D; ++c) { float v = row[c]; ss += v * as[c]; sn += v * an[c]; }
    e_s[i] = ss; e_n[i] = sn;
}

// one block (128 thr) per (head, i) row: leaky_relu + mask + softmax -> bf16 attn
__global__ void k_gat_softmax(const float* __restrict__ e_s, const float* __restrict__ e_n,
                              const float* __restrict__ adj, bf16_t* __restrict__ attn) {
    __shared__ float red[128];
    int h = blockIdx.x / N_NODES;
    int i = blockIdx.x % N_NODES;
    int tid = threadIdx.x;
    float es = e_s[i * HEADS + h];
    float x[4];
#pragma unroll
    for (int k = 0; k < 4; ++k) {
        int j = tid + k * 128;
        float v = es + e_n[j * HEADS + h];
        v = v > 0.0f ? v : 0.2f * v;
        if (adj[(long)i * N_NODES + j] == 0.0f) v = -1.0e9f;
        x[k] = v;
    }
    float mx = fmaxf(fmaxf(x[0], x[1]), fmaxf(x[2], x[3]));
    red[tid] = mx; __syncthreads();
    for (int s = 64; s > 0; s >>= 1) {
        if (tid < s) red[tid] = fmaxf(red[tid], red[tid + s]);
        __syncthreads();
    }
    mx = red[0]; __syncthreads();
    float e[4], sm = 0.0f;
#pragma unroll
    for (int k = 0; k < 4; ++k) { e[k] = __expf(x[k] - mx); sm += e[k]; }
    red[tid] = sm; __syncthreads();
    for (int s = 64; s > 0; s >>= 1) {
        if (tid < s) red[tid] += red[tid + s];
        __syncthreads();
    }
    float inv = 1.0f / red[0];
    bf16_t* row = attn + ((long)h * N_NODES + i) * N_NODES;
#pragma unroll
    for (int k = 0; k < 4; ++k) row[tid + k * 128] = to_bf16(e[k] * inv);
}

// xpT[h][c][j] = bf16(xp[j][h*D + c])
__global__ void k_xp_transpose(const float* __restrict__ xp, bf16_t* __restrict__ xpT) {
    long i = (long)blockIdx.x * blockDim.x + threadIdx.x;
    if (i >= (long)HEADS * D * N_NODES) return;
    int h = (int)(i / (D * N_NODES));
    int r = (int)(i % (D * N_NODES));
    int c = r / N_NODES, j = r % N_NODES;
    xpT[i] = to_bf16(xp[(long)j * (HEADS * D) + h * D + c]);
}

__global__ void k_init_bias_rows(const float* bias, float* out, int rows) {
    int i = blockIdx.x * blockDim.x + threadIdx.x;
    if (i >= rows * D) return;
    out[i] = bias[i & 127];
}

// node_inputs[p][s][c] = bf16(node_state[node_at[p][s]][c]) where valid, else 0
__global__ void k_fill_node_inputs(const float* __restrict__ ns, const int* __restrict__ node_at,
                                   const int* __restrict__ lens, bf16_t* __restrict__ ni) {
    long i = (long)blockIdx.x * blockDim.x + threadIdx.x;
    if (i >= (long)N_PATHS * MAX_LEN * D) return;
    int p = (int)(i >> 10);              // / (MAX_LEN*D)
    int r = (int)(i & 1023);
    int s = r >> 7, c = r & 127;
    float v = 0.0f;
    if (s < lens[p]) v = ns[(long)node_at[p * MAX_LEN + s] * D + c];
    ni[i] = to_bf16(v);
}

// masked GRU step: h := where(mask, hnew, h); accumulate hnew into m2[node]
__global__ void k_gru_step(const float* __restrict__ g1, const float* __restrict__ g2,
                           float* __restrict__ h, const int* __restrict__ lens,
                           const int* __restrict__ node_at, float* __restrict__ m2,
                           int t_actual) {
    long i = (long)blockIdx.x * blockDim.x + threadIdx.x;
    if (i >= (long)N_PATHS * D) return;
    int p = (int)(i >> 7), c = (int)(i & 127);
    if (t_actual >= lens[p]) return;
    const float* a = g1 + (long)p * G3;
    const float* b = g2 + (long)p * G3;
    float z  = sigmoidf(a[c]       + b[c]);
    float rr = sigmoidf(a[D + c]   + b[D + c]);
    float hc = tanhf(a[2 * D + c] + rr * b[2 * D + c]);
    float hold = h[i];
    float hnew = z * hold + (1.0f - z) * hc;
    h[i] = hnew;
    atomicAdd(&m2[(long)node_at[p * MAX_LEN + t_actual] * D + c], hnew);
}

__global__ void k_node_gru(const float* __restrict__ g1, const float* __restrict__ g2,
                           const float* __restrict__ h_in, float* __restrict__ h_out) {
    int i = blockIdx.x * blockDim.x + threadIdx.x;
    if (i >= N_NODES * D) return;
    int n = i >> 7, c = i & 127;
    const float* a = g1 + (long)n * G3;
    const float* b = g2 + (long)n * G3;
    float z  = sigmoidf(a[c]       + b[c]);
    float rr = sigmoidf(a[D + c]   + b[D + c]);
    float hc = tanhf(a[2 * D + c] + rr * b[2 * D + c]);
    h_out[i] = z * h_in[i] + (1.0f - z) * hc;
}

__global__ void k_final_dot(const float* __restrict__ h2, const float* __restrict__ w,
                            const float* __restrict__ b, float* __restrict__ out) {
    int p = blockIdx.x * blockDim.x + threadIdx.x;
    if (p >= N_PATHS) return;
    float s = b[0];
    const float* row = h2 + (long)p * R;
    for (int r = 0; r < R; ++r) s += row[r] * w[r];
    out[p] = s;
}

// ---------------------------------------------------------------------------
// Host side
// ---------------------------------------------------------------------------
static inline void gemm(hipStream_t st, const bf16_t* A, long lda, long sA,
                        const bf16_t* B, long ldb, long sB, const float* bias,
                        float* C, long ldc, long sC, int M, int N, int K,
                        int act, int atomicMode, float alpha, int batch) {
    int tiles = (M / 32) * (N / 32);       // 32x32 output per wave
    dim3 grid((tiles + 7) / 8, 1, batch);
    wmma_gemm_bf16<<<grid, 256, 0, st>>>(A, lda, sA, B, ldb, sB, bias,
                                         C, ldc, sC, M, N, K, act, atomicMode, alpha);
}

#define LAUNCH(kern, n, ...) kern<<<(int)(((long)(n) + 255) / 256), 256, 0, stream>>>(__VA_ARGS__)

extern "C" void kernel_launch(void* const* d_in, const int* in_sizes, int n_in,
                              void* d_out, int out_size, void* d_ws, size_t ws_size,
                              hipStream_t stream) {
    // ---- inputs (setup_inputs order) ----
    const float* ToS      = (const float*)d_in[0];
    const float* bandwith = (const float*)d_in[1];
    const float* q_policy = (const float*)d_in[2];
    const float* w1       = (const float*)d_in[3];
    const float* w2       = (const float*)d_in[4];
    const float* w3       = (const float*)d_in[5];
    const float* queue_sz = (const float*)d_in[6];
    const float* link_cap = (const float*)d_in[7];
    const float* gat_W    = (const float*)d_in[8];
    const float* gat_as   = (const float*)d_in[9];
    const float* gat_an   = (const float*)d_in[10];
    const float* gat_b    = (const float*)d_in[11];
    const float* Wp       = (const float*)d_in[12];
    const float* Up       = (const float*)d_in[13];
    const float* bp       = (const float*)d_in[14];
    const float* Wb       = (const float*)d_in[15];
    const float* Ub       = (const float*)d_in[16];
    const float* bb       = (const float*)d_in[17];
    const float* Wn       = (const float*)d_in[18];
    const float* Un       = (const float*)d_in[19];
    const float* bn       = (const float*)d_in[20];
    const float* rW1      = (const float*)d_in[21];
    const float* rb1      = (const float*)d_in[22];
    const float* rW2      = (const float*)d_in[23];
    const float* rb2      = (const float*)d_in[24];
    const float* rW3      = (const float*)d_in[25];
    const float* rb3      = (const float*)d_in[26];
    const int*   paths    = (const int*)d_in[27];
    const int*   seqs     = (const int*)d_in[28];
    const int*   nodes    = (const int*)d_in[29];
    const int*   adj_idx  = (const int*)d_in[30];
    const int nsteps = in_sizes[27];
    float* out = (float*)d_out;

    // ---- workspace bump allocator ----
    char* wsp = (char*)d_ws;
    auto alloc = [&](size_t bytes) -> void* {
        void* p = wsp;
        wsp += (bytes + 255) & ~(size_t)255;
        return p;
    };
    float* ns_a   = (float*)alloc((size_t)N_NODES * D * 4);
    float* ns_b   = (float*)alloc((size_t)N_NODES * D * 4);
    bf16_t* ns_bf = (bf16_t*)alloc((size_t)N_NODES * D * 2);
    float* m2     = (float*)alloc((size_t)N_NODES * D * 4);
    bf16_t* m2_bf = (bf16_t*)alloc((size_t)N_NODES * D * 2);
    float* adj    = (float*)alloc((size_t)N_NODES * N_NODES * 4);
    float* ssq    = (float*)alloc(256);
    float* xp     = (float*)alloc((size_t)N_NODES * HEADS * D * 4);
    float* e_s    = (float*)alloc((size_t)N_NODES * HEADS * 4);
    float* e_n    = (float*)alloc((size_t)N_NODES * HEADS * 4);
    bf16_t* attn  = (bf16_t*)alloc((size_t)HEADS * N_NODES * N_NODES * 2);
    bf16_t* xpT   = (bf16_t*)alloc((size_t)HEADS * D * N_NODES * 2);
    int* node_at  = (int*)alloc((size_t)N_PATHS * MAX_LEN * 4);
    int* lens     = (int*)alloc((size_t)N_PATHS * 4);
    bf16_t* ni_bf = (bf16_t*)alloc((size_t)N_PATHS * MAX_LEN * D * 2);
    float* pstate = (float*)alloc((size_t)N_PATHS * D * 4);
    float* h_b    = (float*)alloc((size_t)N_PATHS * D * 4);
    bf16_t* h_bf  = (bf16_t*)alloc((size_t)N_PATHS * D * 2);
    float* g1     = (float*)alloc((size_t)N_PATHS * G3 * 4);
    float* g2     = (float*)alloc((size_t)N_PATHS * G3 * 4);
    float* ng1    = (float*)alloc((size_t)N_NODES * G3 * 4);
    float* ng2    = (float*)alloc((size_t)N_NODES * G3 * 4);
    bf16_t* gatWT = (bf16_t*)alloc((size_t)HEADS * D * D * 2);   // [3072][128]
    bf16_t* WpT   = (bf16_t*)alloc((size_t)G3 * D * 2);
    bf16_t* UpT   = (bf16_t*)alloc((size_t)G3 * D * 2);
    bf16_t* WbT   = (bf16_t*)alloc((size_t)G3 * D * 2);
    bf16_t* UbT   = (bf16_t*)alloc((size_t)G3 * D * 2);
    bf16_t* WnT   = (bf16_t*)alloc((size_t)G3 * D * 2);
    bf16_t* UnT   = (bf16_t*)alloc((size_t)G3 * D * 2);
    bf16_t* rW1T  = (bf16_t*)alloc((size_t)R * D * 2);
    bf16_t* rW2T  = (bf16_t*)alloc((size_t)R * R * 2);
    // readout aliases (free after main loop)
    float*  r1  = g1;               // 20000 x 256 f32 (<= 20000 x 384)
    float*  r2  = g2;
    bf16_t* r1b = ni_bf;            // 20000 x 256 bf16 (<= 20000x8x128)

    // ---- setup ----
    LAUNCH(k_build_node_state, N_NODES * D, q_policy, w1, w2, w3, queue_sz, ns_a);
    LAUNCH(k_build_path_state, (long)N_PATHS * D, bandwith, ToS, pstate);
    LAUNCH(k_zero_f32, (long)N_NODES * N_NODES, adj, (long)N_NODES * N_NODES);
    k_reduce_ssq<<<1, 256, 0, stream>>>(link_cap, ssq, N_LINKS);
    LAUNCH(k_scatter_adj, N_LINKS, adj_idx, link_cap, ssq, adj, N_LINKS);
    LAUNCH(k_zero_i32, N_PATHS, lens, (long)N_PATHS);
    LAUNCH(k_count_paths, nsteps, paths, seqs, nodes, lens, node_at, nsteps);

    // weight transposes/conversions: [K][N] f32 -> [N][K] bf16
    LAUNCH(k_transpose_bf16, (long)D * HEADS * D, gat_W, gatWT, D, HEADS * D);
    LAUNCH(k_transpose_bf16, (long)D * G3, Wp, WpT, D, G3);
    LAUNCH(k_transpose_bf16, (long)D * G3, Up, UpT, D, G3);
    LAUNCH(k_transpose_bf16, (long)D * G3, Wb, WbT, D, G3);
    LAUNCH(k_transpose_bf16, (long)D * G3, Ub, UbT, D, G3);
    LAUNCH(k_transpose_bf16, (long)D * G3, Wn, WnT, D, G3);
    LAUNCH(k_transpose_bf16, (long)D * G3, Un, UnT, D, G3);
    LAUNCH(k_transpose_bf16, (long)D * R, rW1, rW1T, D, R);
    LAUNCH(k_transpose_bf16, (long)R * R, rW2, rW2T, R, R);

    // ---- T_STEPS message-passing iterations ----
    for (int step = 0; step < T_STEPS; ++step) {
        // GAT: xp = node_state @ gat_W  (512x128 @ 128x3072)
        LAUNCH(k_f32_to_bf16, (long)N_NODES * D, ns_a, ns_bf, (long)N_NODES * D);
        gemm(stream, ns_bf, D, 0, gatWT, D, 0, nullptr, xp, HEADS * D, 0,
             N_NODES, HEADS * D, D, ACT_NONE, 0, 1.0f, 1);
        LAUNCH(k_es_en, N_NODES * HEADS, xp, gat_as, gat_an, e_s, e_n);
        k_gat_softmax<<<HEADS * N_NODES, 128, 0, stream>>>(e_s, e_n, adj, attn);
        LAUNCH(k_xp_transpose, (long)HEADS * D * N_NODES, xp, xpT);
        LAUNCH(k_init_bias_rows, N_NODES * D, gat_b, ns_b, N_NODES);
        // per-head 512x512 @ 512x128, mean over heads via scaled atomic add
        gemm(stream, attn, N_NODES, (long)N_NODES * N_NODES,
             xpT, N_NODES, (long)D * N_NODES, nullptr,
             ns_b, D, 0, N_NODES, D, N_NODES, ACT_NONE, 1, 1.0f / HEADS, HEADS);

        // gather GAT output into path sequences (bf16 A matrix for gate GEMMs)
        LAUNCH(k_fill_node_inputs, (long)N_PATHS * MAX_LEN * D, ns_b, node_at, lens, ni_bf);
        LAUNCH(k_zero_f32, (long)N_NODES * D, m2, (long)N_NODES * D);
        hipMemcpyAsync(h_b, pstate, (size_t)N_PATHS * D * 4, hipMemcpyDeviceToDevice, stream);

        // forward masked GRU (h lives in pstate; final h becomes next path_state)
        for (int t = 0; t < MAX_LEN; ++t) {
            LAUNCH(k_f32_to_bf16, (long)N_PATHS * D, pstate, h_bf, (long)N_PATHS * D);
            gemm(stream, ni_bf + (long)t * D, (long)MAX_LEN * D, 0, WpT, D, 0, bp,
                 g1, G3, 0, N_PATHS, G3, D, ACT_NONE, 0, 1.0f, 1);
            gemm(stream, h_bf, D, 0, UpT, D, 0, bp + G3,
                 g2, G3, 0, N_PATHS, G3, D, ACT_NONE, 0, 1.0f, 1);
            LAUNCH(k_gru_step, (long)N_PATHS * D, g1, g2, pstate, lens, node_at, m2, t);
        }
        // backward masked GRU (h lives in h_b)
        for (int rt = 0; rt < MAX_LEN; ++rt) {
            int t = MAX_LEN - 1 - rt;
            LAUNCH(k_f32_to_bf16, (long)N_PATHS * D, h_b, h_bf, (long)N_PATHS * D);
            gemm(stream, ni_bf + (long)t * D, (long)MAX_LEN * D, 0, WbT, D, 0, bb,
                 g1, G3, 0, N_PATHS, G3, D, ACT_NONE, 0, 1.0f, 1);
            gemm(stream, h_bf, D, 0, UbT, D, 0, bb + G3,
                 g2, G3, 0, N_PATHS, G3, D, ACT_NONE, 0, 1.0f, 1);
            LAUNCH(k_gru_step, (long)N_PATHS * D, g1, g2, h_b, lens, node_at, m2, t);
        }

        // node GRU: node_state = GRU(m2, node_state)
        LAUNCH(k_f32_to_bf16, (long)N_NODES * D, m2, m2_bf, (long)N_NODES * D);
        LAUNCH(k_f32_to_bf16, (long)N_NODES * D, ns_b, ns_bf, (long)N_NODES * D);
        gemm(stream, m2_bf, D, 0, WnT, D, 0, bn, ng1, G3, 0,
             N_NODES, G3, D, ACT_NONE, 0, 1.0f, 1);
        gemm(stream, ns_bf, D, 0, UnT, D, 0, bn + G3, ng2, G3, 0,
             N_NODES, G3, D, ACT_NONE, 0, 1.0f, 1);
        LAUNCH(k_node_gru, N_NODES * D, ng1, ng2, ns_b, ns_a);
    }

    // ---- readout MLP ----
    LAUNCH(k_f32_to_bf16, (long)N_PATHS * D, pstate, h_bf, (long)N_PATHS * D);
    gemm(stream, h_bf, D, 0, rW1T, D, 0, rb1, r1, R, 0,
         N_PATHS, R, D, ACT_SELU, 0, 1.0f, 1);
    LAUNCH(k_f32_to_bf16, (long)N_PATHS * R, r1, r1b, (long)N_PATHS * R);
    gemm(stream, r1b, R, 0, rW2T, R, 0, rb2, r2, R, 0,
         N_PATHS, R, R, ACT_RELU, 0, 1.0f, 1);
    LAUNCH(k_final_dot, N_PATHS, r2, rW3, rb3, out);

    (void)n_in; (void)out_size; (void)ws_size;
}
